// InducedSelfAttentionBlock_10247791968426
// MI455X (gfx1250) — matrix operand order repaired and checked by reading, not compiled, not run on hardware
//
#include <hip/hip_runtime.h>
#include <cstdint>

typedef __attribute__((ext_vector_type(16))) _Float16 v16h;
typedef __attribute__((ext_vector_type(8)))  _Float16 v8h;
typedef __attribute__((ext_vector_type(4)))  _Float16 v4h;
typedef __attribute__((ext_vector_type(8)))  float    v8f;

union F8  { v8f v; float f[8]; };
union H16 { v16h v; v8h h[2]; _Float16 e[16]; };
union H4  { v4h v; _Float16 e[4]; };

// ---------------------------------------------------------------------------
// CDNA5 async global->LDS copy (ASYNCcnt-tracked, bypasses VGPRs).
// LDS byte address = low 32 bits of the generic pointer (flat LDS aperture).
// ---------------------------------------------------------------------------
__device__ __forceinline__ void async_ld_b128(_Float16* lds, const _Float16* g) {
    uint32_t l = (uint32_t)(uintptr_t)lds;
    asm volatile("global_load_async_to_lds_b128 %0, %1, off"
                 :: "v"(l), "v"(g) : "memory");
}
__device__ __forceinline__ void wait_async0() {
    asm volatile("s_wait_asynccnt 0x0" ::: "memory");
}

// ---------------------------------------------------------------------------
// WMMA fragment loaders (CDNA5 16-bit layouts, wave32)
// A-matrix 16x32 (MxK): lane m=lane&15, K chunk = (lane>>4)*8 and +16
// ---------------------------------------------------------------------------
__device__ __forceinline__ v16h load_frag_a(const _Float16* base, int ld, int lane) {
    int m  = lane & 15;
    int ko = (lane >> 4) << 3;          // 0 or 8
    const _Float16* p = base + m * ld + ko;
    H16 u;
    u.h[0] = *(const v8h*)(p);
    u.h[1] = *(const v8h*)(p + 16);
    return u.v;
}
// B-matrix 32x16 (KxN): lane n=lane&15 holds col n, K chunk = (lane>>4)*8 and +16
__device__ __forceinline__ v16h load_frag_b(const _Float16* base, int ld, int lane) {
    int n  = lane & 15;
    int kb = (lane >> 4) << 3;          // 0 or 8
    const _Float16* p = base + n * ld + kb;
    H16 u;
    u.h[0] = *(const v8h*)(p);
    u.h[1] = *(const v8h*)(p + 16);
    return u.v;
}

__device__ __forceinline__ v8f wmma_f16(v16h a, v16h b, v8f c) {
    return __builtin_amdgcn_wmma_f32_16x16x32_f16(false, a, false, b, (short)0, c,
                                                  false, false);
}

// ---------------------------------------------------------------------------
// fp32 -> fp16 convert
// ---------------------------------------------------------------------------
__global__ void __launch_bounds__(256) cvt_f32_f16(const float* __restrict__ in,
                                                   _Float16* __restrict__ out, int n) {
    int i = blockIdx.x * 256 + threadIdx.x;
    if (i < n) out[i] = (_Float16)in[i];
}

// ---------------------------------------------------------------------------
// LayerNorm over D=512, output f16.  grid = rows, block = 128
// ---------------------------------------------------------------------------
__global__ void __launch_bounds__(128) ln_f16_kernel(const float* __restrict__ x,
                                                     const float* __restrict__ g,
                                                     const float* __restrict__ bta,
                                                     _Float16* __restrict__ out) {
    long row = blockIdx.x;
    int  t   = threadIdx.x;                  // 0..127, 4 elements each
    const float4* xr = (const float4*)(x + row * 512);
    float4 v = xr[t];
    float s = v.x + v.y + v.z + v.w;
    float q = v.x * v.x + v.y * v.y + v.z * v.z + v.w * v.w;
    #pragma unroll
    for (int o = 16; o > 0; o >>= 1) { s += __shfl_xor(s, o); q += __shfl_xor(q, o); }
    __shared__ float ss[4], sq[4];
    int w = t >> 5;
    if ((t & 31) == 0) { ss[w] = s; sq[w] = q; }
    __syncthreads();
    s = ss[0] + ss[1] + ss[2] + ss[3];
    q = sq[0] + sq[1] + sq[2] + sq[3];
    float mean = s * (1.0f / 512.0f);
    float var  = q * (1.0f / 512.0f) - mean * mean;
    float rstd = rsqrtf(var + 1e-5f);
    float4 gv = ((const float4*)g)[t];
    float4 bv = ((const float4*)bta)[t];
    _Float16* orow = out + row * 512 + t * 4;
    orow[0] = (_Float16)((v.x - mean) * rstd * gv.x + bv.x);
    orow[1] = (_Float16)((v.y - mean) * rstd * gv.y + bv.y);
    orow[2] = (_Float16)((v.z - mean) * rstd * gv.z + bv.z);
    orow[3] = (_Float16)((v.w - mean) * rstd * gv.w + bv.w);
}

// ---------------------------------------------------------------------------
// Tiled WMMA GEMM:  C[rows x cols] = A[rows x K] @ W[cols x K]^T + bias
//                   (+ optional exact GELU, + optional residual, f32 or f16 out)
// grid = (cols/128, rows/128), block = 256 (8 waves, each 64x32 of the C tile)
// Double-buffered async global->LDS staging: tile t+1 streams in while tile t
// feeds the WMMA pipe; one barrier per K-step.
// ---------------------------------------------------------------------------
__global__ void __launch_bounds__(256) gemm_wmma(
    const _Float16* __restrict__ A, int lda,
    const _Float16* __restrict__ W, int ldw,
    const float* __restrict__ bias,
    const float* __restrict__ res, int res_mod,
    float* __restrict__ outf, _Float16* __restrict__ outh, int ldc,
    int K, int do_gelu) {

    __shared__ alignas(32) _Float16 sA[2][128 * 32];
    __shared__ alignas(32) _Float16 sB[2][128 * 32];

    int tid  = threadIdx.x;
    int lane = tid & 31;
    int wid  = tid >> 5;
    int wr   = wid >> 2;                 // 0..1  (64-row strip)
    int wc   = wid & 3;                  // 0..3  (32-col strip)
    long rowBase = (long)blockIdx.y * 128;
    long colBase = (long)blockIdx.x * 128;

    F8 acc[4][2];
    #pragma unroll
    for (int i = 0; i < 4; i++)
        #pragma unroll
        for (int j = 0; j < 2; j++)
            #pragma unroll
            for (int e = 0; e < 8; e++) acc[i][j].f[e] = 0.0f;

    auto stage = [&](int k0, int buf) {
        #pragma unroll
        for (int c = tid; c < 512; c += 256) {           // A tile: 128x32 halves
            int r = c >> 2, ko = (c & 3) << 3;
            async_ld_b128(&sA[buf][r * 32 + ko],
                          A + (rowBase + r) * (long)lda + k0 + ko);
        }
        #pragma unroll
        for (int c = tid; c < 512; c += 256) {           // W tile: 128x32 halves
            int r = c >> 2, ko = (c & 3) << 3;
            async_ld_b128(&sB[buf][r * 32 + ko],
                          W + (colBase + r) * (long)ldw + k0 + ko);
        }
    };

    int T = K >> 5;                      // K-steps of 32
    stage(0, 0);
    for (int t = 0; t < T; ++t) {
        wait_async0();                   // this wave's tile-t group landed in LDS
        __syncthreads();                 // all waves' tile t ready; all done with t-1
        if (t + 1 < T) stage((t + 1) << 5, (t + 1) & 1);
        const _Float16* cA = sA[t & 1];
        const _Float16* cB = sB[t & 1];
        v16h af[4], bf[2];
        #pragma unroll
        for (int i = 0; i < 4; i++) af[i] = load_frag_a(&cA[(wr * 64 + i * 16) * 32], 32, lane);
        #pragma unroll
        for (int j = 0; j < 2; j++) bf[j] = load_frag_b(&cB[(wc * 32 + j * 16) * 32], 32, lane);
        #pragma unroll
        for (int i = 0; i < 4; i++)
            #pragma unroll
            for (int j = 0; j < 2; j++)
                acc[i][j].v = wmma_f16(af[i], bf[j], acc[i][j].v);
    }

    int lh = lane >> 4, ln = lane & 15;
    #pragma unroll
    for (int i = 0; i < 4; i++) {
        #pragma unroll
        for (int j = 0; j < 2; j++) {
            long col = colBase + wc * 32 + j * 16 + ln;
            float bv = bias ? bias[col] : 0.0f;
            #pragma unroll
            for (int v = 0; v < 8; v++) {
                long row = rowBase + wr * 64 + i * 16 + v + 8 * lh;
                float val = acc[i][j].f[v] + bv;
                if (do_gelu) val = 0.5f * val * (1.0f + erff(val * 0.70710678118654752f));
                if (res) val += res[(row % (long)res_mod) * ldc + col];
                if (outf) outf[row * ldc + col] = val;
                else      outh[row * ldc + col] = (_Float16)val;
            }
        }
    }
}

// ---------------------------------------------------------------------------
// Flash attention (head dim 64).  One workgroup per (b, head, 128-query chunk).
// 8 waves, each owns a 16-query strip; online softmax over kv tiles of 128.
// K tile staged via async global->LDS; V transposed through VGPRs.
// grid = B*8*nqChunks, block = 256
// ---------------------------------------------------------------------------
__global__ void __launch_bounds__(256) attn_wmma(
    const _Float16* __restrict__ Q, long qBStride,
    const _Float16* __restrict__ Kp, const _Float16* __restrict__ Vp,
    long kvBStride, int ldkv, int kvLen,
    _Float16* __restrict__ O, int oRowsPerBatch,
    int nqChunks, float scale) {

    __shared__ alignas(32) _Float16 sK[128 * 64];   // keys x dims
    __shared__ alignas(32) _Float16 sVt[64 * 128];  // dims x keys (transposed)
    __shared__ alignas(32) _Float16 sP[8 * 16 * 128];

    int bidx = blockIdx.x;
    int qc = bidx % nqChunks;
    int bh = bidx / nqChunks;
    int h  = bh & 7;
    int b  = bh >> 3;

    int tid  = threadIdx.x;
    int lane = tid & 31;
    int w    = tid >> 5;
    int lh   = lane >> 4, ln = lane & 15;

    const _Float16* Qbase = Q + (long)b * qBStride + (long)(qc * 128 + w * 16) * 512 + h * 64;
    v16h qf0 = load_frag_a(Qbase, 512, lane);
    v16h qf1 = load_frag_a(Qbase + 32, 512, lane);

    F8 oaccs[4];
    float mrow[8], lrow[8];
    #pragma unroll
    for (int c = 0; c < 4; c++)
        #pragma unroll
        for (int e = 0; e < 8; e++) oaccs[c].f[e] = 0.0f;
    #pragma unroll
    for (int e = 0; e < 8; e++) { mrow[e] = -1e30f; lrow[e] = 0.0f; }

    const _Float16* Kb = Kp + (long)b * kvBStride + h * 64;
    const _Float16* Vb = Vp + (long)b * kvBStride + h * 64;
    _Float16* pbase = &sP[w * 2048];

    for (int t0 = 0; t0 < kvLen; t0 += 128) {
        // --- stage K tile (row-major) via async copy ---
        #pragma unroll
        for (int c = tid; c < 1024; c += 256) {
            int key = c >> 3, d = (c & 7) << 3;
            async_ld_b128(&sK[key * 64 + d], Kb + (long)(t0 + key) * ldkv + d);
        }
        // --- stage V tile transposed (dims x keys) ---
        #pragma unroll
        for (int c = tid; c < 2048; c += 256) {
            int linear = c << 2;
            int key = linear >> 6, d = linear & 63;
            H4 u; u.v = *(const v4h*)(Vb + (long)(t0 + key) * ldkv + d);
            sVt[(d + 0) * 128 + key] = u.e[0];
            sVt[(d + 1) * 128 + key] = u.e[1];
            sVt[(d + 2) * 128 + key] = u.e[2];
            sVt[(d + 3) * 128 + key] = u.e[3];
        }
        wait_async0();
        __syncthreads();

        // --- S strip = Q(16x64) @ K_tile^T (64x128), scaled ---
        F8 s[8];
        #pragma unroll
        for (int j = 0; j < 8; j++) {
            #pragma unroll
            for (int e = 0; e < 8; e++) s[j].f[e] = 0.0f;
            v16h b0 = load_frag_b(&sK[(j * 16) * 64 + 0], 64, lane);
            v16h b1 = load_frag_b(&sK[(j * 16) * 64 + 32], 64, lane);
            s[j].v = wmma_f16(qf0, b0, s[j].v);
            s[j].v = wmma_f16(qf1, b1, s[j].v);
        }

        // --- online softmax update (rows live across 16-lane halves) ---
        #pragma unroll
        for (int v = 0; v < 8; v++) {
            float tmax = -1e30f;
            #pragma unroll
            for (int j = 0; j < 8; j++) {
                s[j].f[v] *= scale;
                tmax = fmaxf(tmax, s[j].f[v]);
            }
            #pragma unroll
            for (int o = 1; o < 16; o <<= 1) tmax = fmaxf(tmax, __shfl_xor(tmax, o));
            float mnew  = fmaxf(mrow[v], tmax);
            float alpha = __expf(mrow[v] - mnew);
            float rs = 0.0f;
            #pragma unroll
            for (int j = 0; j < 8; j++) {
                float pv = __expf(s[j].f[v] - mnew);
                s[j].f[v] = pv;
                rs += pv;
            }
            #pragma unroll
            for (int o = 1; o < 16; o <<= 1) rs += __shfl_xor(rs, o);
            lrow[v] = lrow[v] * alpha + rs;
            mrow[v] = mnew;
            #pragma unroll
            for (int c = 0; c < 4; c++) oaccs[c].f[v] *= alpha;
        }

        // --- spill P (f16) to per-wave LDS region for A-fragment reload ---
        #pragma unroll
        for (int j = 0; j < 8; j++)
            #pragma unroll
            for (int v = 0; v < 8; v++)
                pbase[(v + 8 * lh) * 128 + j * 16 + ln] = (_Float16)s[j].f[v];

        // --- O(16x64) += P(16x128) @ V_tile(128x64) ---
        #pragma unroll
        for (int c = 0; c < 4; c++) {
            #pragma unroll
            for (int ks = 0; ks < 4; ks++) {
                v16h ap = load_frag_a(pbase + ks * 32, 128, lane);
                v16h bv = load_frag_b(&sVt[(c * 16) * 128 + ks * 32], 128, lane);
                oaccs[c].v = wmma_f16(ap, bv, oaccs[c].v);
            }
        }
        __syncthreads();
    }

    long orow0 = (long)b * oRowsPerBatch + qc * 128 + w * 16;
    #pragma unroll
    for (int c = 0; c < 4; c++)
        #pragma unroll
        for (int v = 0; v < 8; v++)
            O[(orow0 + v + 8 * lh) * 512 + h * 64 + c * 16 + ln] =
                (_Float16)(oaccs[c].f[v] / lrow[v]);
}

// ---------------------------------------------------------------------------
// Host-side orchestration
// ---------------------------------------------------------------------------
extern "C" void kernel_launch(void* const* d_in, const int* in_sizes, int n_in,
                              void* d_out, int out_size, void* d_ws, size_t ws_size,
                              hipStream_t stream) {
    (void)in_sizes; (void)n_in; (void)out_size; (void)ws_size;
    const int Bb = 8, Nn = 4096, Mm = 128;
    const long BN = (long)Bb * Nn;      // 32768
    const long BM = (long)Bb * Mm;      // 1024

    const float* x        = (const float*)d_in[0];
    const float* ind      = (const float*)d_in[1];
    const float* a1_in_w  = (const float*)d_in[2];
    const float* a1_in_b  = (const float*)d_in[3];
    const float* a1_out_w = (const float*)d_in[4];
    const float* a1_out_b = (const float*)d_in[5];
    const float* a1_ln1_g = (const float*)d_in[6];
    const float* a1_ln1_b = (const float*)d_in[7];
    const float* a1_ln2_g = (const float*)d_in[8];
    const float* a1_ln2_b = (const float*)d_in[9];
    const float* a1_l1_w  = (const float*)d_in[10];
    const float* a1_l1_b  = (const float*)d_in[11];
    const float* a1_l2_w  = (const float*)d_in[12];
    const float* a1_l2_b  = (const float*)d_in[13];
    const float* a2_in_w  = (const float*)d_in[14];
    const float* a2_in_b  = (const float*)d_in[15];
    const float* a2_out_w = (const float*)d_in[16];
    const float* a2_out_b = (const float*)d_in[17];
    const float* a2_ln1_g = (const float*)d_in[18];
    const float* a2_ln1_b = (const float*)d_in[19];
    const float* a2_ln2_g = (const float*)d_in[20];
    const float* a2_ln2_b = (const float*)d_in[21];
    const float* a2_l1_w  = (const float*)d_in[22];
    const float* a2_l1_b  = (const float*)d_in[23];
    const float* a2_l2_w  = (const float*)d_in[24];
    const float* a2_l2_b  = (const float*)d_in[25];
    float* out = (float*)d_out;

    char* p = (char*)d_ws;
    auto carve = [&](size_t bytes) -> char* {
        char* r = p;
        p += (bytes + 255) & ~(size_t)255;
        return r;
    };
    _Float16* w_in1  = (_Float16*)carve((size_t)1536 * 512 * 2);
    _Float16* w_out1 = (_Float16*)carve((size_t)512 * 512 * 2);
    _Float16* w_l11  = (_Float16*)carve((size_t)2048 * 512 * 2);
    _Float16* w_l21  = (_Float16*)carve((size_t)512 * 2048 * 2);
    _Float16* w_in2  = (_Float16*)carve((size_t)1536 * 512 * 2);
    _Float16* w_out2 = (_Float16*)carve((size_t)512 * 512 * 2);
    _Float16* w_l12  = (_Float16*)carve((size_t)2048 * 512 * 2);
    _Float16* w_l22  = (_Float16*)carve((size_t)512 * 2048 * 2);
    _Float16* bufA   = (_Float16*)carve((size_t)BN * 512 * 2);   // xn1 / xq2n / x2n
    _Float16* bufKV  = (_Float16*)carve((size_t)BN * 1024 * 2);  // kv1 / (Qh2|O2) / ffh2
    _Float16* qn     = (_Float16*)carve((size_t)Mm * 512 * 2);
    _Float16* qh1    = (_Float16*)carve((size_t)Mm * 512 * 2);
    _Float16* o1     = (_Float16*)carve((size_t)BM * 512 * 2);
    _Float16* x1n    = (_Float16*)carve((size_t)BM * 512 * 2);
    _Float16* ff1h   = (_Float16*)carve((size_t)BM * 2048 * 2);
    _Float16* hn     = (_Float16*)carve((size_t)BM * 512 * 2);
    _Float16* kv2    = (_Float16*)carve((size_t)BM * 1024 * 2);
    float*    x1f    = (float*)carve((size_t)BM * 512 * 4);
    float*    hf     = (float*)carve((size_t)BM * 512 * 4);
    float*    x2f    = (float*)carve((size_t)BN * 512 * 4);
    _Float16* qh2 = bufKV;
    _Float16* o2  = bufKV + (size_t)BN * 512;
    const int BIGMOD = 1 << 30;

    // weight conversion fp32 -> fp16
    auto cvt = [&](const float* s, _Float16* d, int n) {
        cvt_f32_f16<<<(n + 255) / 256, 256, 0, stream>>>(s, d, n);
    };
    cvt(a1_in_w,  w_in1,  1536 * 512);
    cvt(a1_out_w, w_out1, 512 * 512);
    cvt(a1_l1_w,  w_l11,  2048 * 512);
    cvt(a1_l2_w,  w_l21,  512 * 2048);
    cvt(a2_in_w,  w_in2,  1536 * 512);
    cvt(a2_out_w, w_out2, 512 * 512);
    cvt(a2_l1_w,  w_l12,  2048 * 512);
    cvt(a2_l2_w,  w_l22,  512 * 2048);

    // ---------------- MAB1: h = MAB(ind_b, x, x) ----------------
    ln_f16_kernel<<<(int)BN, 128, 0, stream>>>(x,   a1_ln1_g, a1_ln1_b, bufA);
    ln_f16_kernel<<<Mm,      128, 0, stream>>>(ind, a1_ln1_g, a1_ln1_b, qn);
    // Q = LN(ind) @ Wq^T + bq  (shared across batch)
    gemm_wmma<<<dim3(4, 1), 256, 0, stream>>>(qn, 512, w_in1, 512, a1_in_b,
        nullptr, 1, nullptr, qh1, 512, 512, 0);
    // K|V = LN(x) @ [Wk;Wv]^T + [bk;bv]
    gemm_wmma<<<dim3(8, (int)(BN / 128)), 256, 0, stream>>>(bufA, 512,
        w_in1 + (size_t)512 * 512, 512, a1_in_b + 512,
        nullptr, 1, nullptr, bufKV, 1024, 512, 0);
    // flash attention: 128 queries vs 4096 keys per (b,h)
    attn_wmma<<<Bb * 8, 256, 0, stream>>>(qh1, 0L, bufKV, bufKV + 512,
        (long)Nn * 1024, 1024, Nn, o1, Mm, 1, 0.125f);
    // x1 = ind + O @ Wo^T + bo   (residual broadcast over batch: row % 128)
    gemm_wmma<<<dim3(4, (int)(BM / 128)), 256, 0, stream>>>(o1, 512, w_out1, 512,
        a1_out_b, ind, 128, x1f, nullptr, 512, 512, 0);
    ln_f16_kernel<<<(int)BM, 128, 0, stream>>>(x1f, a1_ln2_g, a1_ln2_b, x1n);
    gemm_wmma<<<dim3(16, (int)(BM / 128)), 256, 0, stream>>>(x1n, 512, w_l11, 512,
        a1_l1_b, nullptr, 1, nullptr, ff1h, 2048, 512, 1);
    gemm_wmma<<<dim3(4, (int)(BM / 128)), 256, 0, stream>>>(ff1h, 2048, w_l21, 2048,
        a1_l2_b, x1f, BIGMOD, hf, nullptr, 512, 2048, 0);

    // ---------------- MAB2: out = MAB(x, h, h) ----------------
    ln_f16_kernel<<<(int)BN, 128, 0, stream>>>(x,  a2_ln1_g, a2_ln1_b, bufA);
    ln_f16_kernel<<<(int)BM, 128, 0, stream>>>(hf, a2_ln1_g, a2_ln1_b, hn);
    gemm_wmma<<<dim3(4, (int)(BN / 128)), 256, 0, stream>>>(bufA, 512, w_in2, 512,
        a2_in_b, nullptr, 1, nullptr, qh2, 512, 512, 0);
    gemm_wmma<<<dim3(8, (int)(BM / 128)), 256, 0, stream>>>(hn, 512,
        w_in2 + (size_t)512 * 512, 512, a2_in_b + 512,
        nullptr, 1, nullptr, kv2, 1024, 512, 0);
    // attention: 4096 queries vs 128 keys per (b,h); single kv tile
    attn_wmma<<<Bb * 8 * (Nn / 128), 256, 0, stream>>>(qh2, (long)Nn * 512,
        kv2, kv2 + 512, (long)Mm * 1024, 1024, Mm, o2, Nn, Nn / 128, 0.125f);
    // x2 = x + O @ Wo^T + bo
    gemm_wmma<<<dim3(4, (int)(BN / 128)), 256, 0, stream>>>(o2, 512, w_out2, 512,
        a2_out_b, x, BIGMOD, x2f, nullptr, 512, 512, 0);
    ln_f16_kernel<<<(int)BN, 128, 0, stream>>>(x2f, a2_ln2_g, a2_ln2_b, bufA);
    // FFN split into two 1024-wide hidden halves (bufKV reused as hidden buffer)
    gemm_wmma<<<dim3(8, (int)(BN / 128)), 256, 0, stream>>>(bufA, 512, w_l12, 512,
        a2_l1_b, nullptr, 1, nullptr, bufKV, 1024, 512, 1);
    gemm_wmma<<<dim3(4, (int)(BN / 128)), 256, 0, stream>>>(bufKV, 1024, w_l22, 2048,
        a2_l2_b, x2f, BIGMOD, out, nullptr, 512, 1024, 0);
    gemm_wmma<<<dim3(8, (int)(BN / 128)), 256, 0, stream>>>(bufA, 512,
        w_l12 + (size_t)1024 * 512, 512, a2_l1_b + 1024,
        nullptr, 1, nullptr, bufKV, 1024, 512, 1);
    gemm_wmma<<<dim3(4, (int)(BN / 128)), 256, 0, stream>>>(bufKV, 1024,
        w_l22 + 1024, 2048, nullptr, out, BIGMOD, out, nullptr, 512, 1024, 0);
}
